// GatedSelfAttention_16320875724889
// MI455X (gfx1250) — matrix operand order repaired and checked by reading, not compiled, not run on hardware
//
#include <hip/hip_runtime.h>
#include <hip/hip_bf16.h>
#include <math.h>

// ---------------------------------------------------------------------------
// Types
// ---------------------------------------------------------------------------
typedef __bf16 bf16_t;
typedef __attribute__((ext_vector_type(16))) __bf16 v16bf;
typedef __attribute__((ext_vector_type(8)))  __bf16 v8bf;
typedef __attribute__((ext_vector_type(8)))  float  v8f;
typedef __attribute__((ext_vector_type(4)))  unsigned int u32x4;
typedef __attribute__((ext_vector_type(8)))  int i32x8;
typedef __attribute__((ext_vector_type(4)))  int i32x4;

#define LSEQ   2048
#define TOKENS 4096      // B*L
#define HDIM   2560
#define NH     16
#define NKV    4
#define DHEAD  256
#define INNER  4096      // NH*DHEAD
#define QGN    8192      // 2*INNER
#define KVN    1024      // NKV*DHEAD

#if __has_builtin(__builtin_amdgcn_tensor_load_to_lds)
#define HAS_TDM 1
#endif
#if __has_builtin(__builtin_amdgcn_s_wait_tensorcnt)
#define WAIT_TENSOR() __builtin_amdgcn_s_wait_tensorcnt(0)
#else
#define WAIT_TENSOR() asm volatile("s_wait_tensorcnt 0" ::: "memory")
#endif

union FragU { v16bf v; v8bf h[2]; };

// A-matrix 16x32 bf16 fragment (ISA 7.12.2): lane half 0 holds K 0-7 (v0-3)
// and 16-23 (v4-7); half 1 holds K 8-15 and 24-31. rowptr = &row[kbase].
__device__ inline v16bf ld_frag_a(const bf16_t* rowptr, int half) {
  FragU f;
  f.h[0] = *(const v8bf*)(rowptr + half * 8);
  f.h[1] = *(const v8bf*)(rowptr + 16 + half * 8);
  return f.v;
}
// B-matrix 32x16 bf16 fragment (same K order as A): half0: K 0-7, 16-23;
// half1: K 8-15, 24-31; col = lane&15. colptr = &T[n][kbase] with K contiguous.
__device__ inline v16bf ld_frag_b(const bf16_t* colptr, int half) {
  FragU f;
  f.h[0] = *(const v8bf*)(colptr + half * 8);
  f.h[1] = *(const v8bf*)(colptr + 16 + half * 8);
  return f.v;
}
__device__ inline v8f wmma_bf16(v16bf a, v16bf b, v8f c) {
  return __builtin_amdgcn_wmma_f32_16x16x32_bf16(
      false, a, false, b, (short)0, c, false, false);
}
__device__ inline float red_sum16(float x) {
  x += __shfl_xor(x, 1, 32); x += __shfl_xor(x, 2, 32);
  x += __shfl_xor(x, 4, 32); x += __shfl_xor(x, 8, 32);
  return x;
}
__device__ inline float red_max16(float x) {
  x = fmaxf(x, __shfl_xor(x, 1, 32)); x = fmaxf(x, __shfl_xor(x, 2, 32));
  x = fmaxf(x, __shfl_xor(x, 4, 32)); x = fmaxf(x, __shfl_xor(x, 8, 32));
  return x;
}
// Async global->LDS copy of 16B (per active lane), tracked by ASYNCcnt.
__device__ inline void async_b128(const void* gptr, void* lptr) {
  asm volatile("global_load_async_to_lds_b128 %0, %1, off"
               :: "v"((unsigned)(size_t)lptr),
                  "v"((unsigned long long)(size_t)gptr)
               : "memory");
}
__device__ inline void wait_async0() {
  asm volatile("s_wait_asynccnt 0" ::: "memory");
}

// ---------------------------------------------------------------------------
// f32 -> bf16 conversion (grid-stride)
// ---------------------------------------------------------------------------
__global__ __launch_bounds__(256)
void cvt_f32_bf16(const float* __restrict__ s, bf16_t* __restrict__ d, size_t n) {
  size_t i = (size_t)blockIdx.x * blockDim.x + threadIdx.x;
  size_t stride = (size_t)gridDim.x * blockDim.x;
  for (; i < n; i += stride) d[i] = (bf16_t)s[i];
}

// ---------------------------------------------------------------------------
// Generic bf16 WMMA GEMM: C[M,N](f32) = A[M,K](bf16 row-major) @ B[K,N](bf16)
// BM=BN=128, BK=64, 256 threads = 8 waves (2x4 wave grid, 64x32 per wave).
// A tile staged via global_load_async_to_lds_b128 (ASYNCcnt).
// ---------------------------------------------------------------------------
#define BM 128
#define BN 128
#define BK 64

__global__ __launch_bounds__(256)
void gemm_bf16f32(const bf16_t* __restrict__ A, const bf16_t* __restrict__ B,
                  float* __restrict__ C, int K, int lda, int ldb, int ldc) {
  __shared__ bf16_t At[BM][BK + 8];   // k contiguous
  __shared__ bf16_t Bt[BN][BK + 8];   // transposed: k contiguous
  const int tid  = threadIdx.x;
  const int lane = tid & 31;
  const int wid  = tid >> 5;
  const int half = lane >> 4;
  const int l16  = lane & 15;
  const int bm0  = blockIdx.x * BM;
  const int bn0  = blockIdx.y * BN;
  const int wm   = (wid >> 2) * 64;   // 0 / 64
  const int wn   = (wid & 3) * 32;    // 0..96

  v8f acc[4][2];
  v8f zf = {};
#pragma unroll
  for (int mi = 0; mi < 4; ++mi)
#pragma unroll
    for (int ni = 0; ni < 2; ++ni) acc[mi][ni] = zf;

  for (int k0 = 0; k0 < K; k0 += BK) {
    // --- stage A tile (128x64) asynchronously into LDS ---
#pragma unroll
    for (int r = 0; r < 4; ++r) {
      int idx = (tid + r * 256) * 8;        // 0..8191
      int row = idx >> 6;
      int col = idx & 63;
      async_b128(A + (size_t)(bm0 + row) * lda + k0 + col, &At[row][col]);
    }
    // --- stage B tile transposed (Bt[n][k]) ---
#pragma unroll
    for (int r = 0; r < 4; ++r) {
      int task = tid + r * 256;             // 0..1023
      int kk = task >> 4;                   // 0..63
      int nc = (task & 15) * 8;
      v8bf val = *(const v8bf*)(B + (size_t)(k0 + kk) * ldb + bn0 + nc);
#pragma unroll
      for (int e = 0; e < 8; ++e) Bt[nc + e][kk] = val[e];
    }
    wait_async0();
    __syncthreads();

#pragma unroll
    for (int ks = 0; ks < 2; ++ks) {
      v16bf af[4], bfr[2];
#pragma unroll
      for (int mi = 0; mi < 4; ++mi)
        af[mi] = ld_frag_a(&At[wm + mi * 16 + l16][ks * 32], half);
#pragma unroll
      for (int ni = 0; ni < 2; ++ni)
        bfr[ni] = ld_frag_b(&Bt[wn + ni * 16 + l16][ks * 32], half);
#pragma unroll
      for (int mi = 0; mi < 4; ++mi)
#pragma unroll
        for (int ni = 0; ni < 2; ++ni)
          acc[mi][ni] = wmma_bf16(af[mi], bfr[ni], acc[mi][ni]);
    }
    __syncthreads();
  }

  // epilogue: C layout — VGPR i: row = i + 8*half, col = lane&15
  float* cbase = C + (size_t)(bm0 + wm + half * 8) * ldc + bn0 + wn + l16;
#pragma unroll
  for (int mi = 0; mi < 4; ++mi)
#pragma unroll
    for (int i = 0; i < 8; ++i) {
      float* cp = cbase + (size_t)(mi * 16 + i) * ldc;
      cp[0]  = acc[mi][0][i];
      cp[16] = acc[mi][1][i];
    }
}

// ---------------------------------------------------------------------------
// Fused RMSNorm + RoPE; one wave per (token, head) row of 256; bf16 output
// in [b][h][l][d] layout.
// ---------------------------------------------------------------------------
__global__ __launch_bounds__(256)
void rmsnorm_rope(const float* __restrict__ X, int src_ld,
                  const float* __restrict__ w,
                  const float* __restrict__ cosb, const float* __restrict__ sinb,
                  bf16_t* __restrict__ out, int nheads) {
  int gw   = blockIdx.x * 8 + (threadIdx.x >> 5);
  int lane = threadIdx.x & 31;
  int t = gw / nheads;           // token = b*L + l
  int h = gw % nheads;
  int l = t & (LSEQ - 1);
  const float* src = X + (size_t)t * src_ld + h * DHEAD;
  int d0 = lane * 8;

  float x[8];
  float ss = 0.0f;
#pragma unroll
  for (int e = 0; e < 8; ++e) { x[e] = src[d0 + e]; ss += x[e] * x[e]; }
  ss = red_sum16(ss);
  ss += __shfl_xor(ss, 16, 32);              // full 32-lane reduction
  float rms = rsqrtf(ss * (1.0f / 256.0f) + 1e-6f);
#pragma unroll
  for (int e = 0; e < 8; ++e) x[e] = x[e] * rms * w[d0 + e];

  // rope: d<128 pairs with d+128 (lane ^ 16)
  float sgn = (lane < 16) ? -1.0f : 1.0f;
  const float* cs = cosb + (size_t)l * DHEAD;
  const float* sn = sinb + (size_t)l * DHEAD;
  v8bf ov;
#pragma unroll
  for (int e = 0; e < 8; ++e) {
    float rot = sgn * __shfl_xor(x[e], 16, 32);
    ov[e] = (bf16_t)(x[e] * cs[d0 + e] + rot * sn[d0 + e]);
  }
  bf16_t* dst = out + ((size_t)(((t / LSEQ) * nheads + h) * LSEQ + l)) * DHEAD + d0;
  *(v8bf*)dst = ov;
}

// ---------------------------------------------------------------------------
// V relayout: [token][kv*256+d] f32 -> [b][kv][l][d] bf16
// ---------------------------------------------------------------------------
__global__ __launch_bounds__(256)
void v_relayout(const float* __restrict__ vf, bf16_t* __restrict__ vo) {
  size_t i = (size_t)blockIdx.x * blockDim.x + threadIdx.x;
  if (i >= (size_t)TOKENS * KVN) return;
  int t  = (int)(i >> 10);
  int c  = (int)(i & 1023);
  int b  = t / LSEQ, l = t & (LSEQ - 1);
  int kv = c >> 8,   d = c & 255;
  vo[(((size_t)(b * NKV + kv)) * LSEQ + l) * DHEAD + d] = (bf16_t)vf[i];
}

// ---------------------------------------------------------------------------
// Flash attention: BM=BN=64, D=256, 128 threads (4 waves, 16 q-rows each).
// K tile staged by the Tensor Data Mover (tensor_load_to_lds + TENSORcnt),
// V^T staged manually (transpose), Q frags cached in registers, online
// softmax with 16-lane shfl reductions, P via per-wave LDS to A-layout.
// ---------------------------------------------------------------------------
#define FA_BM 64
#define FA_BN 64

__global__ __launch_bounds__(128)
void flash_attn(const bf16_t* __restrict__ Q, const bf16_t* __restrict__ Kc,
                const bf16_t* __restrict__ Vc, float* __restrict__ O) {
  __shared__ bf16_t Kt[FA_BN][DHEAD + 8];      // [kv][d]; row = 512B + 16B pad
  __shared__ bf16_t Vt[DHEAD][FA_BN + 8];      // [d][kv]  (transposed)
  __shared__ bf16_t Pst[4][16][FA_BN + 8];     // per-wave P staging

  const int tid  = threadIdx.x;
  const int lane = tid & 31;
  const int wid  = tid >> 5;
  const int half = lane >> 4;
  const int l16  = lane & 15;
  const int q0   = blockIdx.x * FA_BM;
  const int bh   = blockIdx.y;                 // b*NH + h
  const int b    = bh >> 4, h = bh & 15, kvh = h >> 2;

  const bf16_t* qrow  = Q + ((size_t)bh * LSEQ + q0 + wid * 16 + l16) * DHEAD;
  const bf16_t* kbase = Kc + (size_t)(b * NKV + kvh) * LSEQ * DHEAD;
  const bf16_t* vbase = Vc + (size_t)(b * NKV + kvh) * LSEQ * DHEAD;

  // cache all 8 K-step Q fragments in registers (gfx1250: up to 1024 VGPRs)
  v16bf qf[8];
#pragma unroll
  for (int ks = 0; ks < 8; ++ks) qf[ks] = ld_frag_a(qrow + ks * 32, half);

  v8f zf = {};
  v8f o[16];
#pragma unroll
  for (int t = 0; t < 16; ++t) o[t] = zf;
  float mrun[8], lrun[8];
#pragma unroll
  for (int i = 0; i < 8; ++i) { mrun[i] = -3.0e38f; lrun[i] = 0.0f; }

  const int jmax = q0 / FA_BN;                 // inclusive (causal)
  for (int j = 0; j <= jmax; ++j) {
    const int kv0 = j * FA_BN;

#ifdef HAS_TDM
    // ---- K tile via Tensor Data Mover: 2D tile 256 x 64 rows, bf16,
    //      LDS padding 4 DWORDs every 128 DWORDs (=> [64][264] layout) ----
    if (wid == 0) {
      unsigned long long ga = (unsigned long long)(size_t)(kbase + (size_t)kv0 * DHEAD);
      u32x4 g0 = { 1u,                                   // count=1 (user D#)
                   (unsigned)(size_t)(void*)&Kt[0][0],   // lds_addr
                   (unsigned)ga,                         // global_addr[31:0]
                   (unsigned)(ga >> 32) | (2u << 30) };  // ga[56:32] | type=2
      i32x8 g1 = { (1 << 16) | (1 << 20) | (6 << 22) | (3 << 25),
                   //^ data_size=2B, pad_enable, pad_interval=128DW, pad=4DW
                   (int)(256u << 16),                    // tensor_dim0 = 256
                   (int)(((unsigned)(LSEQ - kv0) & 0xFFFFu) << 16), // tensor_dim1
                   (int)(256u << 16),                    // tile_dim0 = 256
                   64,                                   // tile_dim1 = 64
                   256,                                  // tensor_dim0_stride
                   0, 0 };
      i32x4 g2 = { 0, 0, 0, 0 };
      i32x4 g3 = { 0, 0, 0, 0 };
#if __clang_major__ >= 23
      i32x8 gz = {};
      __builtin_amdgcn_tensor_load_to_lds(g0, g1, g2, g3, gz, 0);
#else
      __builtin_amdgcn_tensor_load_to_lds(g0, g1, g2, g3, 0);
#endif
    }
#endif
    // ---- V^T tile staged by all waves (transpose scatter) ----
#pragma unroll
    for (int r = 0; r < 16; ++r) {
      int idx = (tid + r * 128) * 8;           // 0..16376
      int row = idx >> 8;                      // kv row 0..63
      int col = idx & 255;                     // d
      v8bf vv8 = *(const v8bf*)(vbase + (size_t)(kv0 + row) * DHEAD + col);
#pragma unroll
      for (int e = 0; e < 8; ++e) Vt[col + e][row] = vv8[e];
#ifndef HAS_TDM
      v8bf kv8 = *(const v8bf*)(kbase + (size_t)(kv0 + row) * DHEAD + col);
      *(v8bf*)(&Kt[row][col]) = kv8;
#endif
    }
#ifdef HAS_TDM
    if (wid == 0) WAIT_TENSOR();
#endif
    __syncthreads();

    // ---- S = Q K^T  (16x64 per wave) ----
    v8f s[4];
#pragma unroll
    for (int nt = 0; nt < 4; ++nt) s[nt] = zf;
#pragma unroll
    for (int ks = 0; ks < 8; ++ks) {
#pragma unroll
      for (int nt = 0; nt < 4; ++nt) {
        v16bf bfr = ld_frag_b(&Kt[nt * 16 + l16][ks * 32], half);
        s[nt] = wmma_bf16(qf[ks], bfr, s[nt]);
      }
    }

    // ---- scale + causal mask ----
    const bool diag = (j == jmax);
#pragma unroll
    for (int nt = 0; nt < 4; ++nt)
#pragma unroll
      for (int i = 0; i < 8; ++i) {
        float val = s[nt][i] * 0.0625f;        // 1/sqrt(256)
        if (diag) {
          int qi = q0 + wid * 16 + half * 8 + i;
          int ki = kv0 + nt * 16 + l16;
          if (ki > qi) val = -3.0e38f;
        }
        s[nt][i] = val;
      }

    // ---- online softmax ----
    float mnew[8], alpha[8], rsum[8];
#pragma unroll
    for (int i = 0; i < 8; ++i) {
      float mx = fmaxf(fmaxf(s[0][i], s[1][i]), fmaxf(s[2][i], s[3][i]));
      mx = red_max16(mx);
      mnew[i]  = fmaxf(mrun[i], mx);
      alpha[i] = __expf(mrun[i] - mnew[i]);
      rsum[i]  = 0.0f;
    }
#pragma unroll
    for (int nt = 0; nt < 4; ++nt)
#pragma unroll
      for (int i = 0; i < 8; ++i) {
        float p = __expf(s[nt][i] - mnew[i]);
        s[nt][i] = p;
        rsum[i] += p;
      }
#pragma unroll
    for (int i = 0; i < 8; ++i) {
      rsum[i] = red_sum16(rsum[i]);
      lrun[i] = lrun[i] * alpha[i] + rsum[i];
      mrun[i] = mnew[i];
    }
#pragma unroll
    for (int t = 0; t < 16; ++t)
#pragma unroll
      for (int i = 0; i < 8; ++i) o[t][i] = o[t][i] * alpha[i];

    // ---- P -> LDS (C-layout -> A-layout round trip, per-wave region) ----
#pragma unroll
    for (int nt = 0; nt < 4; ++nt)
#pragma unroll
      for (int i = 0; i < 8; ++i)
        Pst[wid][half * 8 + i][nt * 16 + l16] = (bf16_t)s[nt][i];

    // ---- O += P V ----
#pragma unroll
    for (int ks = 0; ks < 2; ++ks) {
      v16bf pa = ld_frag_a(&Pst[wid][l16][ks * 32], half);
#pragma unroll
      for (int dt = 0; dt < 16; ++dt) {
        v16bf vb = ld_frag_b(&Vt[dt * 16 + l16][ks * 32], half);
        o[dt] = wmma_bf16(pa, vb, o[dt]);
      }
    }
    __syncthreads();
  }

  // ---- epilogue: O / l, write f32 [token][h*256+d] ----
  float* obase = O + (size_t)(b * LSEQ + q0 + wid * 16 + half * 8) * INNER
                   + h * DHEAD + l16;
#pragma unroll
  for (int i = 0; i < 8; ++i) {
    float inv = 1.0f / lrun[i];
    float* op = obase + (size_t)i * INNER;
#pragma unroll
    for (int dt = 0; dt < 16; ++dt) op[dt * 16] = o[dt][i] * inv;
  }
}

// ---------------------------------------------------------------------------
// Gating: out_bf = attn * silu(gate)
// ---------------------------------------------------------------------------
__global__ __launch_bounds__(256)
void gate_silu(const float* __restrict__ attn, const float* __restrict__ qg,
               bf16_t* __restrict__ out) {
  size_t i = (size_t)blockIdx.x * blockDim.x + threadIdx.x;
  size_t n = (size_t)TOKENS * INNER;
  size_t stride = (size_t)gridDim.x * blockDim.x;
  for (; i < n; i += stride) {
    size_t t = i >> 12;                 // /4096
    size_t c = i & 4095;
    float g = qg[t * QGN + INNER + c];
    float a = attn[i];
    out[i] = (bf16_t)(a * (g / (1.0f + __expf(-g))));
  }
}

// ---------------------------------------------------------------------------
// Host launcher
// ---------------------------------------------------------------------------
extern "C" void kernel_launch(void* const* d_in, const int* in_sizes, int n_in,
                              void* d_out, int out_size, void* d_ws, size_t ws_size,
                              hipStream_t stream) {
  (void)in_sizes; (void)n_in; (void)out_size; (void)ws_size;
  const float* hs   = (const float*)d_in[0];
  const float* cosb = (const float*)d_in[1];
  const float* sinb = (const float*)d_in[2];
  const float* Wq   = (const float*)d_in[3];
  const float* Wk   = (const float*)d_in[4];
  const float* Wv   = (const float*)d_in[5];
  const float* Wo   = (const float*)d_in[6];
  const float* qnw  = (const float*)d_in[7];
  const float* knw  = (const float*)d_in[8];

  char* ws = (char*)d_ws;
  auto alloc = [&](size_t bytes) -> char* {
    char* p = ws; ws += (bytes + 255) & ~(size_t)255; return p;
  };
  bf16_t* hs_bf  = (bf16_t*)alloc((size_t)TOKENS * HDIM * 2);
  bf16_t* wq_bf  = (bf16_t*)alloc((size_t)HDIM * QGN * 2);
  bf16_t* wk_bf  = (bf16_t*)alloc((size_t)HDIM * KVN * 2);
  bf16_t* wv_bf  = (bf16_t*)alloc((size_t)HDIM * KVN * 2);
  bf16_t* wo_bf  = (bf16_t*)alloc((size_t)INNER * HDIM * 2);
  float*  qg_f   = (float*)alloc((size_t)TOKENS * QGN * 4);
  float*  kf     = (float*)alloc((size_t)TOKENS * KVN * 4);
  float*  vf     = (float*)alloc((size_t)TOKENS * KVN * 4);
  bf16_t* q_bf   = (bf16_t*)alloc((size_t)TOKENS * INNER * 2);  // [b][h][l][d]
  bf16_t* k_bf   = (bf16_t*)alloc((size_t)TOKENS * KVN * 2);    // [b][kv][l][d]
  bf16_t* v_bf   = (bf16_t*)alloc((size_t)TOKENS * KVN * 2);
  float*  attn_f = (float*)alloc((size_t)TOKENS * INNER * 4);   // [token][h*d]
  bf16_t* gat_bf = (bf16_t*)alloc((size_t)TOKENS * INNER * 2);

  // 1) convert to bf16
  cvt_f32_bf16<<<2048, 256, 0, stream>>>(hs, hs_bf, (size_t)TOKENS * HDIM);
  cvt_f32_bf16<<<2048, 256, 0, stream>>>(Wq, wq_bf, (size_t)HDIM * QGN);
  cvt_f32_bf16<<<2048, 256, 0, stream>>>(Wk, wk_bf, (size_t)HDIM * KVN);
  cvt_f32_bf16<<<2048, 256, 0, stream>>>(Wv, wv_bf, (size_t)HDIM * KVN);
  cvt_f32_bf16<<<2048, 256, 0, stream>>>(Wo, wo_bf, (size_t)INNER * HDIM);

  // 2) projections
  dim3 gqg(TOKENS / BM, QGN / BN);
  gemm_bf16f32<<<gqg, 256, 0, stream>>>(hs_bf, wq_bf, qg_f, HDIM, HDIM, QGN, QGN);
  dim3 gkv(TOKENS / BM, KVN / BN);
  gemm_bf16f32<<<gkv, 256, 0, stream>>>(hs_bf, wk_bf, kf, HDIM, HDIM, KVN, KVN);
  gemm_bf16f32<<<gkv, 256, 0, stream>>>(hs_bf, wv_bf, vf, HDIM, HDIM, KVN, KVN);

  // 3) RMSNorm + RoPE (q, k) ; relayout v
  rmsnorm_rope<<<(TOKENS * NH) / 8, 256, 0, stream>>>(qg_f, QGN, qnw, cosb, sinb, q_bf, NH);
  rmsnorm_rope<<<(TOKENS * NKV) / 8, 256, 0, stream>>>(kf, KVN, knw, cosb, sinb, k_bf, NKV);
  v_relayout<<<((size_t)TOKENS * KVN + 255) / 256, 256, 0, stream>>>(vf, v_bf);

  // 4) flash attention
  dim3 gfa(LSEQ / FA_BM, 2 * NH);
  flash_attn<<<gfa, 128, 0, stream>>>(q_bf, k_bf, v_bf, attn_f);

  // 5) gate
  gate_silu<<<8192, 256, 0, stream>>>(attn_f, qg_f, gat_bf);

  // 6) output projection -> d_out (f32)
  dim3 gwo(TOKENS / BM, HDIM / BN);
  gemm_bf16f32<<<gwo, 256, 0, stream>>>(gat_bf, wo_bf, (float*)d_out, INNER, INNER, HDIM, HDIM);
}
